// DecoderWithAttention_71829033058663
// MI455X (gfx1250) — compile-verified, matrix operand order checked by reading, not run on hardware
//
#include <hip/hip_runtime.h>

typedef __bf16 bf16_t;
typedef __attribute__((ext_vector_type(16))) __bf16 v16bf;
typedef __attribute__((ext_vector_type(8)))  __bf16 v8bf;
typedef __attribute__((ext_vector_type(8)))  float  v8f;

#define B_    64
#define P_    144
#define E_    512
#define A_    256
#define EMB_  256
#define D_    512
#define V_    2000
#define T_    128
#define XK_   1280          // [emb(256) | gate*awe(512) | h(512)]
#define G4_   2048
#define NB_   32            // persistent blocks in recurrent kernel
#define PRED_ (B_*T_*V_)    // 16,384,000 floats

// ---------------------------------------------------------------------------
// WMMA fragment loaders (wave32, CDNA5 layouts).
// A 16x32 bf16: lane(0..15)=row M, holds K 0..7 & 16..23; lane(16..31)=row M,
// holds K 8..15 & 24..31.  B is loaded identically from the weight matrix
// stored as W[n][k] (n-major, k contiguous) so each lane streams 16B chunks.
// Works for global (global_load_b128) and LDS (ds_load_b128) pointers alike.
// ---------------------------------------------------------------------------
__device__ __forceinline__ v16bf frag_bf16(const bf16_t* base, int ld, int lane) {
  const bf16_t* p = base + (size_t)(lane & 15) * ld + ((lane >> 4) << 3);
  v8bf lo = *(const v8bf*)p;
  v8bf hi = *(const v8bf*)(p + 16);
  v16bf f;
#pragma unroll
  for (int i = 0; i < 8; ++i) { f[i] = lo[i]; f[8 + i] = hi[i]; }
  return f;
}

__device__ __forceinline__ v16bf frag_f32(const float* base, int ld, int lane) {
  const float* p = base + (size_t)(lane & 15) * ld + ((lane >> 4) << 3);
  v16bf f;
#pragma unroll
  for (int i = 0; i < 8; ++i) { f[i] = (bf16_t)p[i]; f[8 + i] = (bf16_t)p[16 + i]; }
  return f;
}

// D(16x16,f32) += A(16x32,bf16) * B(32x16,bf16)  over K
__device__ __forceinline__ v8f gemm_tile_bf16(const bf16_t* Aa, int lda,
                                              const bf16_t* Bb, int ldb,
                                              int K, int lane, v8f acc) {
  for (int k = 0; k < K; k += 32) {
    __builtin_prefetch(Bb + k + 64, 0, 1);          // global_prefetch_b8 (L2 stream)
    v16bf a = frag_bf16(Aa + k, lda, lane);
    v16bf b = frag_bf16(Bb + k, ldb, lane);
    acc = __builtin_amdgcn_wmma_f32_16x16x32_bf16(false, a, false, b,
                                                  (short)0, acc, false, false);
  }
  return acc;
}

__device__ __forceinline__ v8f gemm_tile_f32(const float* Aa, int lda,
                                             const float* Bb, int ldb,
                                             int K, int lane, v8f acc) {
  for (int k = 0; k < K; k += 32) {
    v16bf a = frag_f32(Aa + k, lda, lane);
    v16bf b = frag_f32(Bb + k, ldb, lane);
    acc = __builtin_amdgcn_wmma_f32_16x16x32_bf16(false, a, false, b,
                                                  (short)0, acc, false, false);
  }
  return acc;
}

__device__ __forceinline__ float sigm(float x) { return 1.f / (1.f + expf(-x)); }

// ---------------------------------------------------------------------------
struct Prm {
  // inputs (setup_inputs order)
  const float* enc;  const int* caps;  const int* caplen; const float* emb;
  const float* Wea;  const float* bea; const float* Wda;  const float* bda;
  const float* Wfull;const float* bfull;
  const float* Wih;  const float* bih; const float* Whh;  const float* bhh;
  const float* Wih0; const float* bih0;const float* Wic;  const float* bic;
  const float* Wfb;  const float* bfb; const float* Wfc;  const float* bfc;
  // output
  float* out;
  // workspace
  unsigned* bar; int* sind; int* dlen; int* caps_s;
  float* att2gate; float* gates; float* cst; float* att1; float* ba; float* bg;
  bf16_t* xcat; bf16_t* embbf; bf16_t* Hall; bf16_t* WaB; bf16_t* WgB; bf16_t* WfcB;
};

// ---------------------------------------------------------------------------
// Kernel 0: stable length-sort, caps/dec_len outputs, barrier reset
// ---------------------------------------------------------------------------
__global__ void k_sort(Prm p) {
  const int tid = threadIdx.x;
  if (tid < B_) {
    int lb = p.caplen[tid];
    int rank = 0;
    for (int j = 0; j < B_; ++j) {
      int lj = p.caplen[j];
      if (lj > lb || (lj == lb && j < tid)) ++rank;
    }
    p.sind[rank] = tid;                       // descending, stable
  }
  __syncthreads();
  if (tid < B_) {
    int s  = p.sind[tid];
    int dl = p.caplen[s] - 1;
    p.dlen[tid] = dl;
    p.out[PRED_ + B_ * T_ + tid] = (float)dl;
  }
  __syncthreads();
  for (int idx = tid; idx < B_ * T_; idx += blockDim.x) {
    int b = idx >> 7, t = idx & (T_ - 1);
    int cv = p.caps[p.sind[b] * T_ + t];
    p.caps_s[idx] = cv;
    p.out[PRED_ + idx] = (float)cv;
  }
  if (tid == 0) *p.bar = 0u;
}

// ---------------------------------------------------------------------------
// Kernel 1: precompute.  blocks [0,64): att1 via WMMA;  [64,128): bf16
// weight/embedding conversion;  [128,192): mean_enc -> h0/c0 + emb(t=0).
// ---------------------------------------------------------------------------
__global__ void k_pre(Prm p) {
  const int bid  = blockIdx.x;
  const int tid  = threadIdx.x;
  const int lane = tid & 31;
  __shared__ float meanv[E_];

  if (bid < 64) {
    // att1[b,p,a] = enc[b,p,:] . Wea[a,:] + bea[a]   (M=9216, N=256, K=512)
    const int gw = bid * 8 + (tid >> 5);                 // 512 waves
    for (int tile = gw; tile < 576 * 16; tile += 512) {
      int tm = tile >> 4, tn = tile & 15;
      int b  = (tm * 16) / P_;
      int p0 = tm * 16 - b * P_;                         // 144 % 16 == 0
      const float* Aa = p.enc + ((size_t)p.sind[b] * P_ + p0) * E_;
      const float* Bb = p.Wea + (size_t)(tn * 16) * E_;
      v8f acc = {};
      acc = gemm_tile_f32(Aa, E_, Bb, E_, E_, lane, acc);
      int n = (tn * 16) + (lane & 15);
      float bias = p.bea[n];
#pragma unroll
      for (int i = 0; i < 8; ++i) {
        int m = ((lane >> 4) << 3) + i;
        p.att1[(size_t)(b * P_ + p0 + m) * A_ + n] = acc[i] + bias;
      }
    }
  } else if (bid < 128) {
    // flat bf16 conversions
    const long N1 = (long)G4_ * XK_;        // Wg = [Wih | Whh]
    const long N2 = 768L * D_;              // Wa = [Wda ; Wfb]
    const long N3 = (long)V_ * D_;          // Wfc
    const long N4 = (long)T_ * B_ * EMB_;   // emb gather
    const long N5 = 768, N6 = G4_;
    const long tot = N1 + N2 + N3 + N4 + N5 + N6;
    for (long idx = (long)(bid - 64) * 256 + tid; idx < tot; idx += 64L * 256) {
      long r = idx;
      if (r < N1) {
        int n = (int)(r / XK_), k = (int)(r % XK_);
        float v = (k < 768) ? p.Wih[(size_t)n * 768 + k] : p.Whh[(size_t)n * D_ + (k - 768)];
        p.WgB[r] = (bf16_t)v;
      } else if ((r -= N1) < N2) {
        int n = (int)(r / D_), k = (int)(r % D_);
        float v = (n < A_) ? p.Wda[(size_t)n * D_ + k] : p.Wfb[(size_t)(n - A_) * D_ + k];
        p.WaB[r] = (bf16_t)v;
      } else if ((r -= N2) < N3) {
        p.WfcB[r] = (bf16_t)p.Wfc[r];
      } else if ((r -= N3) < N4) {
        int e = (int)(r % EMB_);
        long bt = r / EMB_;
        int b = (int)(bt % B_), t = (int)(bt / B_);
        p.embbf[r] = (bf16_t)p.emb[(size_t)p.caps_s[b * T_ + t] * EMB_ + e];
      } else if ((r -= N4) < N5) {
        p.ba[r] = (r < A_) ? p.bda[r] : p.bfb[r - A_];
      } else {
        r -= N5;
        p.bg[r] = p.bih[r] + p.bhh[r];
      }
    }
  } else {
    // per-batch init: one block per b
    int b = bid - 128;
    const float* encb = p.enc + (size_t)p.sind[b] * P_ * E_;
    for (int e = tid; e < E_; e += blockDim.x) {
      float s = 0.f;
      for (int q = 0; q < P_; ++q) s += encb[(size_t)q * E_ + e];
      meanv[e] = s * (1.0f / P_);
    }
    __syncthreads();
    for (int j = tid; j < 2 * D_; j += blockDim.x) {
      int d = j & (D_ - 1);
      const float* w = (j < D_) ? (p.Wih0 + (size_t)d * E_) : (p.Wic + (size_t)d * E_);
      float acc = (j < D_) ? p.bih0[d] : p.bic[d];
      for (int e = 0; e < E_; ++e) acc += meanv[e] * w[e];
      if (j < D_) p.xcat[(size_t)b * XK_ + 768 + d] = (bf16_t)acc;   // h0 (bf16 for GEMMs)
      else        p.cst[(size_t)b * D_ + d] = acc;                   // c0 (fp32 state)
    }
    if (tid < EMB_)
      p.xcat[(size_t)b * XK_ + tid] =
          (bf16_t)p.emb[(size_t)p.caps_s[b * T_] * EMB_ + tid];      // emb(t=0)
  }
}

// ---------------------------------------------------------------------------
// grid-wide barrier (sense-free monotonic counter; reset by k_sort)
// ---------------------------------------------------------------------------
__device__ __forceinline__ void gbar(unsigned* bar, unsigned target) {
  __syncthreads();
  __threadfence();                                    // release my stores
  if (threadIdx.x == 0) {
    atomicAdd(bar, 1u);
    while (atomicAdd(bar, 0u) < target) __builtin_amdgcn_s_sleep(2);
  }
  __syncthreads();
  __threadfence();                                    // acquire others' stores
}

// ---------------------------------------------------------------------------
// Kernel 2: persistent recurrent loop (32 blocks x 256 threads = 256 waves)
// ---------------------------------------------------------------------------
__global__ void k_rnn(Prm p) {
  const int tid  = threadIdx.x;
  const int lane = tid & 31;
  const int gw   = blockIdx.x * 8 + (tid >> 5);       // global wave id
  __shared__ float e_s[P_];
  unsigned target = 0;

  for (int t = 0; t < T_; ++t) {
    // ---- Phase A: [att2 | gate_lin] = h @ [Wda;Wfb]^T   (64x768, K=512)
    for (int tile = gw; tile < 4 * 48; tile += NB_ * 8) {
      int tm = tile / 48, tn = tile % 48;
      const bf16_t* Aa = p.xcat + 768 + (size_t)(tm * 16) * XK_;
      const bf16_t* Bb = p.WaB + (size_t)(tn * 16) * D_;
      v8f acc = {};
      acc = gemm_tile_bf16(Aa, XK_, Bb, D_, D_, lane, acc);
      int n = tn * 16 + (lane & 15);
      float bias = p.ba[n];
#pragma unroll
      for (int i = 0; i < 8; ++i) {
        int m = ((lane >> 4) << 3) + i;
        p.att2gate[(size_t)(tm * 16 + m) * 768 + n] = acc[i] + bias;
      }
    }
    target += NB_; gbar(p.bar, target);

    // ---- Phase B: attention scores, softmax, awe, gate  (2 b per block)
    for (int b = blockIdx.x; b < B_; b += NB_) {
      const float* a1 = p.att1 + (size_t)b * P_ * A_;
      const float* a2 = p.att2gate + (size_t)b * 768;
      for (int q = tid; q < P_; q += blockDim.x) {
        float s = 0.f;
        const float* row = a1 + (size_t)q * A_;
        for (int a = 0; a < A_; ++a) {
          float v = row[a] + a2[a];
          s += (v > 0.f ? v : 0.f) * p.Wfull[a];
        }
        e_s[q] = s + p.bfull[0];
      }
      __syncthreads();
      if (tid == 0) {
        float mx = e_s[0];
        for (int q = 1; q < P_; ++q) mx = fmaxf(mx, e_s[q]);
        float sum = 0.f;
        for (int q = 0; q < P_; ++q) { e_s[q] = expf(e_s[q] - mx); sum += e_s[q]; }
        float inv = 1.f / sum;
        for (int q = 0; q < P_; ++q) e_s[q] *= inv;
      }
      __syncthreads();
      const float* encb = p.enc + (size_t)p.sind[b] * P_ * E_;
      for (int d = tid; d < D_; d += blockDim.x) {
        float s = 0.f;
        for (int q = 0; q < P_; ++q) s += e_s[q] * encb[(size_t)q * E_ + d];
        float gate = sigm(a2[A_ + d]);
        p.xcat[(size_t)b * XK_ + EMB_ + d] = (bf16_t)(gate * s);
      }
      __syncthreads();
    }
    target += NB_; gbar(p.bar, target);

    // ---- Phase C: gates = [emb|g*awe|h] @ [Wih|Whh]^T   (64x2048, K=1280)
    for (int tile = gw; tile < 4 * 128; tile += NB_ * 8) {
      int tm = tile >> 7, tn = tile & 127;
      const bf16_t* Aa = p.xcat + (size_t)(tm * 16) * XK_;
      const bf16_t* Bb = p.WgB + (size_t)(tn * 16) * XK_;
      v8f acc = {};
      acc = gemm_tile_bf16(Aa, XK_, Bb, XK_, XK_, lane, acc);
      int n = tn * 16 + (lane & 15);
      float bias = p.bg[n];
#pragma unroll
      for (int i = 0; i < 8; ++i) {
        int m = ((lane >> 4) << 3) + i;
        p.gates[(size_t)(tm * 16 + m) * G4_ + n] = acc[i] + bias;
      }
    }
    target += NB_; gbar(p.bar, target);

    // ---- Phase D: LSTM pointwise + state/emb staging for t+1
    for (int idx = blockIdx.x * blockDim.x + tid; idx < B_ * D_; idx += NB_ * 256) {
      int b = idx >> 9, d = idx & (D_ - 1);
      const float* g = p.gates + (size_t)b * G4_;
      float iv = sigm(g[d]);
      float fv = sigm(g[D_ + d]);
      float gv = tanhf(g[2 * D_ + d]);
      float ov = sigm(g[3 * D_ + d]);
      float cn = fv * p.cst[idx] + iv * gv;
      p.cst[idx] = cn;
      float hn = ov * tanhf(cn);
      bf16_t hb = (bf16_t)hn;
      p.xcat[(size_t)b * XK_ + 768 + d] = hb;                 // h for next step
      p.Hall[((size_t)b * T_ + t) * D_ + d] = hb;             // for final GEMM
    }
    if (t + 1 < T_) {
      for (int idx = blockIdx.x * blockDim.x + tid; idx < B_ * EMB_; idx += NB_ * 256) {
        int b = idx >> 8, e = idx & (EMB_ - 1);
        p.xcat[(size_t)b * XK_ + e] = p.embbf[((size_t)(t + 1) * B_ + b) * EMB_ + e];
      }
    }
    target += NB_; gbar(p.bar, target);
  }
}

// ---------------------------------------------------------------------------
// Kernel 3: preds = Hall(8192x512) @ Wfc^T(512x2000) + bfc, masked.
// One block per 16-row A panel.  The panel (16x512 bf16 = 16KB) is staged
// into LDS with GLOBAL_LOAD_ASYNC_TO_LDS_B128 (ASYNCcnt), drained with
// s_wait_asynccnt; the 8 waves then sweep the 125 vocab tiles reading A
// fragments from LDS (ds_load_b128) and Wfc from L2.
// ---------------------------------------------------------------------------
__global__ void k_preds(Prm p) {
  const int tid  = threadIdx.x;
  const int lane = tid & 31;
  const int wid  = tid >> 5;
  const int NT   = V_ / 16;                       // 125 vocab tiles
  __shared__ bf16_t shA[16 * D_];                 // 16KB A panel

  for (int pm = blockIdx.x; pm < (B_ * T_) / 16; pm += gridDim.x) {
    // ---- async-stage A panel: 16KB = 256 threads x 4 chunks x 16B
    const bf16_t* panel = p.Hall + (size_t)pm * 16 * D_;
    unsigned ldsBase = (unsigned)(size_t)(void*)shA;
#pragma unroll
    for (int j = 0; j < 4; ++j) {
      int cidx = tid + 256 * j;
      unsigned ldsa = ldsBase + cidx * 16;        // per-lane LDS byte address
      unsigned goff = cidx * 16;                  // per-lane global byte offset
      asm volatile("global_load_async_to_lds_b128 %0, %1, %2"
                   :: "v"(ldsa), "v"(goff), "s"(panel) : "memory");
    }
    asm volatile("s_wait_asynccnt 0" ::: "memory");
    __syncthreads();

    // ---- sweep vocab tiles
    for (int tn = wid; tn < NT; tn += 8) {
      const bf16_t* Bb = p.WfcB + (size_t)(tn * 16) * D_;
      v8f acc = {};
      acc = gemm_tile_bf16(shA, D_, Bb, D_, D_, lane, acc);
      int n = tn * 16 + (lane & 15);
      float bias = p.bfc[n];
#pragma unroll
      for (int i = 0; i < 8; ++i) {
        int m = ((lane >> 4) << 3) + i;
        int r = pm * 16 + m;
        int b = r >> 7, t = r & (T_ - 1);
        float v = (t < p.dlen[b]) ? (acc[i] + bias) : 0.0f;
        p.out[(size_t)r * V_ + n] = v;
      }
    }
    __syncthreads();                              // protect shA reuse
  }
}

// ---------------------------------------------------------------------------
extern "C" void kernel_launch(void* const* d_in, const int* in_sizes, int n_in,
                              void* d_out, int out_size, void* d_ws, size_t ws_size,
                              hipStream_t stream) {
  (void)in_sizes; (void)n_in; (void)out_size; (void)ws_size;
  Prm p;
  p.enc    = (const float*)d_in[0];  p.caps  = (const int*)d_in[1];
  p.caplen = (const int*)d_in[2];    p.emb   = (const float*)d_in[3];
  p.Wea = (const float*)d_in[4];  p.bea = (const float*)d_in[5];
  p.Wda = (const float*)d_in[6];  p.bda = (const float*)d_in[7];
  p.Wfull = (const float*)d_in[8]; p.bfull = (const float*)d_in[9];
  p.Wih = (const float*)d_in[10]; p.bih = (const float*)d_in[11];
  p.Whh = (const float*)d_in[12]; p.bhh = (const float*)d_in[13];
  p.Wih0 = (const float*)d_in[14]; p.bih0 = (const float*)d_in[15];
  p.Wic = (const float*)d_in[16]; p.bic = (const float*)d_in[17];
  p.Wfb = (const float*)d_in[18]; p.bfb = (const float*)d_in[19];
  p.Wfc = (const float*)d_in[20]; p.bfc = (const float*)d_in[21];
  p.out = (float*)d_out;

  char* w = (char*)d_ws;
  auto take = [&](size_t bytes) { char* q = w; w += (bytes + 255) & ~(size_t)255; return q; };
  p.bar      = (unsigned*)take(sizeof(unsigned));
  p.sind     = (int*)take(B_ * 4);
  p.dlen     = (int*)take(B_ * 4);
  p.caps_s   = (int*)take((size_t)B_ * T_ * 4);
  p.att2gate = (float*)take((size_t)B_ * 768 * 4);
  p.gates    = (float*)take((size_t)B_ * G4_ * 4);
  p.cst      = (float*)take((size_t)B_ * D_ * 4);
  p.att1     = (float*)take((size_t)B_ * P_ * A_ * 4);
  p.ba       = (float*)take(768 * 4);
  p.bg       = (float*)take(G4_ * 4);
  p.xcat     = (bf16_t*)take((size_t)B_ * XK_ * 2);
  p.embbf    = (bf16_t*)take((size_t)T_ * B_ * EMB_ * 2);
  p.Hall     = (bf16_t*)take((size_t)B_ * T_ * D_ * 2);
  p.WaB      = (bf16_t*)take((size_t)768 * D_ * 2);
  p.WgB      = (bf16_t*)take((size_t)G4_ * XK_ * 2);
  p.WfcB     = (bf16_t*)take((size_t)V_ * D_ * 2);

  hipLaunchKernelGGL(k_sort,  dim3(1),   dim3(256), 0, stream, p);
  hipLaunchKernelGGL(k_pre,   dim3(192), dim3(256), 0, stream, p);
  hipLaunchKernelGGL(k_rnn,   dim3(NB_), dim3(256), 0, stream, p);
  hipLaunchKernelGGL(k_preds, dim3(512), dim3(256), 0, stream, p);
}
